// AdaptiveBertSelfAttention_81020263071744
// MI455X (gfx1250) — matrix operand order repaired and checked by reading, not compiled, not run on hardware
//
#include <hip/hip_runtime.h>
#include <hip/hip_bf16.h>

#define BB 4
#define SS 2048
#define DD 768
#define HH 12
#define HD 64
#define MM (BB*SS)   // 8192

typedef __attribute__((ext_vector_type(16))) __bf16 bf16x16;
typedef __attribute__((ext_vector_type(8)))  __bf16 bf16x8;
typedef __attribute__((ext_vector_type(4)))  __bf16 bf16x4;
typedef __attribute__((ext_vector_type(8)))  float  v8f;

__device__ inline v8f zero8() {
    v8f z = {0.f,0.f,0.f,0.f,0.f,0.f,0.f,0.f};
    return z;
}

// A fragment 16x32 bf16 (MxK), row-major source with leading dim lda.
// lane 0-15 -> M=lane; element e: k=half*16+e
__device__ inline bf16x16 load_a_frag(const __bf16* A, int lda, int m0, int k0, int lane) {
    int m  = m0 + (lane & 15);
    int hf = lane >> 4;
    const __bf16* p = A + (size_t)m * lda + k0 + hf * 16;
    bf16x16 f;
    *(bf16x8*)&f        = *(const bf16x8*)p;          // k = k0+hf*16 .. +7
    *((bf16x8*)&f + 1)  = *(const bf16x8*)(p + 8);    // k = k0+hf*16+8 .. +15
    return f;
}

// B fragment 32x16 bf16 (KxN) where B[k][n] = W[n][k] (W row-major, ld=ldw).
// lane: n = n0 + (lane&15); element e: k = k0 + half*16 + e  (16 contiguous bf16)
__device__ inline bf16x16 load_b_fragT(const __bf16* W, int ldw, int k0, int n0, int lane) {
    int n  = n0 + (lane & 15);
    int hf = lane >> 4;
    return *(const bf16x16*)(W + (size_t)n * ldw + k0 + hf * 16);
}

// ---------------- f32 -> bf16 convert ----------------
__global__ void cvt_f32_bf16(const float* __restrict__ in, __bf16* __restrict__ out, int n) {
    int i = (blockIdx.x * blockDim.x + threadIdx.x) * 4;
    if (i + 3 < n) {
        float4 v = *(const float4*)(in + i);
        bf16x4 o;
        o[0] = (__bf16)v.x; o[1] = (__bf16)v.y; o[2] = (__bf16)v.z; o[3] = (__bf16)v.w;
        *(bf16x4*)(out + i) = o;
    }
}

// ---------------- QKV projection GEMM ----------------
// C[M=8192,N=768] = hs @ W^T + bias ; z=0:Q (scaled), z=1:K, z=2:V (transposed store)
__global__ __launch_bounds__(256) void qkv_gemm(
    const __bf16* __restrict__ hs,
    const __bf16* __restrict__ wq, const __bf16* __restrict__ wk, const __bf16* __restrict__ wv,
    const float* __restrict__ bq, const float* __restrict__ bk, const float* __restrict__ bv,
    const float* __restrict__ scal,
    __bf16* __restrict__ qb, __bf16* __restrict__ kb, __bf16* __restrict__ vtb)
{
    const int z    = blockIdx.z;
    const __bf16* W    = (z == 0) ? wq : ((z == 1) ? wk : wv);
    const float*  bias = (z == 0) ? bq : ((z == 1) ? bk : bv);

    const int lane = threadIdx.x & 31;
    const int wave = threadIdx.x >> 5;
    const int m0   = blockIdx.y * 128 + wave * 16;
    const int n0   = blockIdx.x * 64;

    v8f acc[4];
#pragma unroll
    for (int g = 0; g < 4; ++g) acc[g] = zero8();

    for (int k0 = 0; k0 < DD; k0 += 32) {
        bf16x16 a = load_a_frag(hs, DD, m0, k0, lane);
#pragma unroll
        for (int g = 0; g < 4; ++g) {
            bf16x16 bfr = load_b_fragT(W, DD, k0, n0 + g * 16, lane);
            acc[g] = __builtin_amdgcn_wmma_f32_16x16x32_bf16(
                false, a, false, bfr, (short)0, acc[g], false, false);
        }
    }

    const int hf = lane >> 4;
    const int nn = lane & 15;
#pragma unroll
    for (int g = 0; g < 4; ++g) {
        int n = n0 + g * 16 + nn;
        float bia = bias[n];
        float sc  = (z == 0) ? (scal[n >> 6] * 0.125f) : 1.0f;  // fold 1/sqrt(64)*scale into Q
#pragma unroll
        for (int r = 0; r < 8; ++r) {
            int m = m0 + hf * 8 + r;
            float v = (acc[g][r] + bia) * sc;
            if (z == 2) {
                int b_ = m >> 11, s_ = m & (SS - 1);
                vtb[((size_t)b_ * DD + n) * SS + s_] = (__bf16)v;   // V^T: [B, D, S]
            } else if (z == 1) {
                kb[(size_t)m * DD + n] = (__bf16)v;
            } else {
                qb[(size_t)m * DD + n] = (__bf16)v;
            }
        }
    }
}

// ---------------- flash attention ----------------
// grid: (S/128, H, B); 8 waves, each wave: 16 q-rows, full K loop.
__global__ __launch_bounds__(256) void attn(
    const __bf16* __restrict__ qb, const __bf16* __restrict__ kb,
    const __bf16* __restrict__ vtb, const float* __restrict__ mask,
    float* __restrict__ out)
{
    __shared__ __attribute__((aligned(64))) __bf16 Pl[8][16 * 32]; // per-wave P tile (1KB)

    const int lane = threadIdx.x & 31;
    const int wave = threadIdx.x >> 5;
    const int h    = blockIdx.y;
    const int b    = blockIdx.z;
    const int q0   = blockIdx.x * 128 + wave * 16;
    const int hf   = lane >> 4;
    const int nn   = lane & 15;

    const __bf16* Q  = qb  + (size_t)b * SS * DD;
    const __bf16* K  = kb  + (size_t)b * SS * DD;
    const __bf16* Vt = vtb + (size_t)b * DD * SS;
    const float*  mk = mask + (size_t)b * SS;

    // Q fragments for this wave's 16 rows (head slice h*64 .. h*64+63)
    bf16x16 qa0 = load_a_frag(Q, DD, q0, h * HD + 0,  lane);
    bf16x16 qa1 = load_a_frag(Q, DD, q0, h * HD + 32, lane);

    v8f o[4];
#pragma unroll
    for (int g = 0; g < 4; ++g) o[g] = zero8();
    float mrow[8], lrow[8];
#pragma unroll
    for (int r = 0; r < 8; ++r) { mrow[r] = -1e30f; lrow[r] = 0.f; }

    __bf16* P = &Pl[wave][0];

    for (int j = 0; j < SS; j += 32) {
        // ---- scores: 16 x 32 tile = Q(16x64) x K^T(64x32) ----
        bf16x16 b00 = load_b_fragT(K, DD, h * HD + 0,  j,      lane);
        bf16x16 b01 = load_b_fragT(K, DD, h * HD + 32, j,      lane);
        bf16x16 b10 = load_b_fragT(K, DD, h * HD + 0,  j + 16, lane);
        bf16x16 b11 = load_b_fragT(K, DD, h * HD + 32, j + 16, lane);

        v8f s0 = zero8(), s1 = zero8();
        s0 = __builtin_amdgcn_wmma_f32_16x16x32_bf16(false, qa0, false, b00, (short)0, s0, false, false);
        s0 = __builtin_amdgcn_wmma_f32_16x16x32_bf16(false, qa1, false, b01, (short)0, s0, false, false);
        s1 = __builtin_amdgcn_wmma_f32_16x16x32_bf16(false, qa0, false, b10, (short)0, s1, false, false);
        s1 = __builtin_amdgcn_wmma_f32_16x16x32_bf16(false, qa1, false, b11, (short)0, s1, false, false);

        // additive mask (broadcast over rows; column = key index)
        float mk0 = mk[j + nn];
        float mk1 = mk[j + 16 + nn];
#pragma unroll
        for (int r = 0; r < 8; ++r) { s0[r] += mk0; s1[r] += mk1; }

        // ---- online softmax ----
        float mx[8];
#pragma unroll
        for (int r = 0; r < 8; ++r) mx[r] = fmaxf(s0[r], s1[r]);
#pragma unroll
        for (int d = 1; d < 16; d <<= 1)
#pragma unroll
            for (int r = 0; r < 8; ++r) mx[r] = fmaxf(mx[r], __shfl_xor(mx[r], d, 32));

        float alpha[8];
#pragma unroll
        for (int r = 0; r < 8; ++r) {
            float mn = fmaxf(mrow[r], mx[r]);
            alpha[r] = __expf(mrow[r] - mn);
            mrow[r]  = mn;
        }
#pragma unroll
        for (int r = 0; r < 8; ++r) { s0[r] = __expf(s0[r] - mrow[r]); s1[r] = __expf(s1[r] - mrow[r]); }

        float sm[8];
#pragma unroll
        for (int r = 0; r < 8; ++r) sm[r] = s0[r] + s1[r];
#pragma unroll
        for (int d = 1; d < 16; d <<= 1)
#pragma unroll
            for (int r = 0; r < 8; ++r) sm[r] += __shfl_xor(sm[r], d, 32);
#pragma unroll
        for (int r = 0; r < 8; ++r) lrow[r] = lrow[r] * alpha[r] + sm[r];
#pragma unroll
        for (int g = 0; g < 4; ++g)
#pragma unroll
            for (int r = 0; r < 8; ++r) o[g][r] *= alpha[r];

        // ---- C-layout -> A-layout transpose of P through LDS ----
#pragma unroll
        for (int r = 0; r < 8; ++r) {
            int row = hf * 8 + r;
            P[row * 32 + nn]      = (__bf16)s0[r];
            P[row * 32 + 16 + nn] = (__bf16)s1[r];
        }
        asm volatile("s_wait_dscnt 0" ::: "memory");

        bf16x16 pa;
        {
            const __bf16* pp = P + nn * 32 + hf * 16;  // A-frag: m=lane&15
            *(bf16x8*)&pa       = *(const bf16x8*)pp;
            *((bf16x8*)&pa + 1) = *(const bf16x8*)(pp + 8);
        }
        asm volatile("" ::: "memory");

        // ---- O += P(16x32) x V(32x64), V^T rows are head-dims, contiguous in keys ----
#pragma unroll
        for (int g = 0; g < 4; ++g) {
            bf16x16 vb = load_b_fragT(Vt, SS, /*k0=*/j, /*n0=*/h * HD + g * 16, lane);
            o[g] = __builtin_amdgcn_wmma_f32_16x16x32_bf16(
                false, pa, false, vb, (short)0, o[g], false, false);
        }
    }

    // ---- normalize and store ctx [B,S,D] ----
#pragma unroll
    for (int g = 0; g < 4; ++g) {
#pragma unroll
        for (int r = 0; r < 8; ++r) {
            int m = q0 + hf * 8 + r;
            int n = h * HD + g * 16 + nn;
            out[((size_t)b * SS + m) * DD + n] = o[g][r] / lrow[r];
        }
    }
}

extern "C" void kernel_launch(void* const* d_in, const int* in_sizes, int n_in,
                              void* d_out, int out_size, void* d_ws, size_t ws_size,
                              hipStream_t stream) {
    const float* hs   = (const float*)d_in[0];
    const float* mask = (const float*)d_in[1];
    const float* Wq   = (const float*)d_in[2];
    const float* bq   = (const float*)d_in[3];
    const float* Wk   = (const float*)d_in[4];
    const float* bk   = (const float*)d_in[5];
    const float* Wv   = (const float*)d_in[6];
    const float* bv   = (const float*)d_in[7];
    const float* scal = (const float*)d_in[8];
    float* out = (float*)d_out;

    char* w = (char*)d_ws;
    size_t off = 0;
    auto alloc = [&](size_t bytes) -> void* {
        void* p = w + off;
        off = (off + bytes + 255) & ~(size_t)255;
        return p;
    };
    __bf16* hsb = (__bf16*)alloc((size_t)MM * DD * 2);
    __bf16* wqb = (__bf16*)alloc((size_t)DD * DD * 2);
    __bf16* wkb = (__bf16*)alloc((size_t)DD * DD * 2);
    __bf16* wvb = (__bf16*)alloc((size_t)DD * DD * 2);
    __bf16* qb  = (__bf16*)alloc((size_t)MM * DD * 2);
    __bf16* kb  = (__bf16*)alloc((size_t)MM * DD * 2);
    __bf16* vtb = (__bf16*)alloc((size_t)MM * DD * 2);
    (void)ws_size; (void)in_sizes; (void)n_in; (void)out_size;

    const int nHS = MM * DD;        // 6,291,456
    const int nW  = DD * DD;        // 589,824
    cvt_f32_bf16<<<(nHS / 4 + 255) / 256, 256, 0, stream>>>(hs, hsb, nHS);
    cvt_f32_bf16<<<(nW  / 4 + 255) / 256, 256, 0, stream>>>(Wq, wqb, nW);
    cvt_f32_bf16<<<(nW  / 4 + 255) / 256, 256, 0, stream>>>(Wk, wkb, nW);
    cvt_f32_bf16<<<(nW  / 4 + 255) / 256, 256, 0, stream>>>(Wv, wvb, nW);

    dim3 ggrid(DD / 64, MM / 128, 3);      // (12, 64, 3)
    qkv_gemm<<<ggrid, 256, 0, stream>>>(hsb, wqb, wkb, wvb, bq, bk, bv, scal, qb, kb, vtb);

    dim3 agrid(SS / 128, HH, BB);          // (16, 12, 4)
    attn<<<agrid, 256, 0, stream>>>(qb, kb, vtb, mask, out);
}
